// LongShortAttention_29738353557556
// MI455X (gfx1250) — compile-verified, matrix operand order checked by reading, not run on hardware
//
#include <hip/hip_runtime.h>
#include <hip/hip_bf16.h>

typedef __attribute__((ext_vector_type(16))) _Float16 v16h;
typedef __attribute__((ext_vector_type(8)))  float    v8f;
typedef __attribute__((ext_vector_type(4)))  int      v4i;

#define NSEQ  4096
#define DIMM  1024
#define NHEAD 16
#define DH    64
#define BH    64
#define WWIN  128
#define NSEG  256
#define SSEG  16

#if defined(__has_builtin)
#if __has_builtin(__builtin_amdgcn_global_load_async_to_lds_b128)
#define USE_ASYNC_LDS 1
#endif
#endif

typedef v4i __attribute__((address_space(1)))* gas_v4i;   // global v4i*
typedef v4i __attribute__((address_space(3)))* las_v4i;   // LDS v4i*

// copy 16 f16 (32B) global -> LDS; async path uses 2x b128 async-to-LDS ops
static __device__ inline void copy16h_g2l(const _Float16* gsrc, _Float16* ldst){
#ifdef USE_ASYNC_LDS
  __builtin_amdgcn_global_load_async_to_lds_b128(
      (gas_v4i)(unsigned long long)gsrc,
      (las_v4i)(unsigned)(unsigned long long)ldst, 0, 0);
  __builtin_amdgcn_global_load_async_to_lds_b128(
      (gas_v4i)(unsigned long long)(gsrc + 8),
      (las_v4i)(unsigned)(unsigned long long)(ldst + 8), 0, 0);
#else
  const unsigned* s32 = (const unsigned*)gsrc;
  unsigned* d32 = (unsigned*)ldst;
  #pragma unroll
  for (int i = 0; i < 8; ++i) d32[i] = s32[i];
#endif
}
static __device__ inline void async_wait(){
#ifdef USE_ASYNC_LDS
#if __has_builtin(__builtin_amdgcn_s_wait_asynccnt)
  __builtin_amdgcn_s_wait_asynccnt(0);
#else
  asm volatile("s_wait_asynccnt 0x0" ::: "memory");
#endif
#endif
}

static __device__ inline unsigned short f2s(_Float16 x){ union{_Float16 f; unsigned short s;} u; u.f=x; return u.s; }
static __device__ inline unsigned pack2(_Float16 lo, _Float16 hi){ return (unsigned)f2s(lo) | ((unsigned)f2s(hi)<<16); }

// A-style fragment (also B when source stored [n][k] row-major): 16 rows x 32 k
template<int LDA>
static __device__ inline v16h load_a_lds(const _Float16* base, int rbase, int kb){
  int lane = threadIdx.x & 31;
  int m = lane & 15, koff = (lane >> 4) * 8;
  union { v16h v; unsigned u[8]; } r;
  const _Float16* row = base + (size_t)(rbase + m) * LDA + kb;
  #pragma unroll
  for (int v = 0; v < 8; ++v){
    int k = (v < 4) ? (koff + 2*v) : (16 + koff + 2*(v-4));
    r.u[v] = *(const unsigned*)(row + k);
  }
  return r.v;
}

// B fragment when source stored [k][n] row-major (stride LDA)
template<int LDA>
static __device__ inline v16h load_b_kmajor(const _Float16* base, int nbase, int kb){
  int lane = threadIdx.x & 31;
  int n = nbase + (lane & 15), koff = (lane >> 4) * 8;
  union { v16h v; unsigned u[8]; } r;
  #pragma unroll
  for (int v = 0; v < 8; ++v){
    int k = (v < 4) ? (koff + 2*v) : (16 + koff + 2*(v-4));
    r.u[v] = pack2(base[(size_t)(kb + k) * LDA + n], base[(size_t)(kb + k + 1) * LDA + n]);
  }
  return r.v;
}

// ---------------- Kernel 1: QKV projection + rotary ----------------
__global__ __launch_bounds__(128) void qkv_rope_kernel(
    const float* __restrict__ x, const float* __restrict__ Wq, const float* __restrict__ Wkv,
    _Float16* __restrict__ qout, _Float16* __restrict__ kvout)
{
  const int rowTile = blockIdx.x;   // 256 tiles of 64 rows over B*N
  const int head    = blockIdx.y;   // 16
  const int isKV    = blockIdx.z;   // 0=q 1=kv
  const float* Wsel = isKV ? Wkv : Wq;
  _Float16* outb    = isKV ? kvout : qout;

  __shared__ _Float16 As[64][72];   // [row][k]
  __shared__ _Float16 Bs[64][72];   // [n][k]

  const int t = threadIdx.x, wave = t >> 5, lane = t & 31;
  const int row0 = rowTile * 64, colb = head * DH;

  v8f acc[4] = {};
  for (int kk = 0; kk < DIMM; kk += 64){
    {
      int r = t >> 1, c0 = (t & 1) * 32;
      const float* src = x + (size_t)(row0 + r) * DIMM + kk + c0;
      #pragma unroll
      for (int i = 0; i < 32; ++i) As[r][c0 + i] = (_Float16)src[i];
      int kr = t >> 1, n0 = (t & 1) * 32;
      const float* wsrc = Wsel + (size_t)(kk + kr) * DIMM + colb + n0;
      #pragma unroll
      for (int i = 0; i < 32; ++i) Bs[n0 + i][kr] = (_Float16)wsrc[i];
    }
    __syncthreads();
    #pragma unroll
    for (int ks = 0; ks < 2; ++ks){
      v16h a = load_a_lds<72>(&As[0][0], wave * 16, ks * 32);
      #pragma unroll
      for (int nb = 0; nb < 4; ++nb){
        v16h b = load_a_lds<72>(&Bs[0][0], nb * 16, ks * 32);
        acc[nb] = __builtin_amdgcn_wmma_f32_16x16x32_f16(false, a, false, b, (short)0, acc[nb], false, false);
      }
    }
    __syncthreads();
  }
  // rotary epilogue: pairs (d, d+32) live in accs nb and nb+2, same lane
  const int mbase = 8 * (lane >> 4);
  #pragma unroll
  for (int nb = 0; nb < 2; ++nb){
    int d = nb * 16 + (lane & 15);                       // 0..31
    // inv_freq = 10000^(-2d/64) = exp2(-d * log2(10000)/32)
    float invf = exp2f((float)d * -0.4152410118609203f);
    #pragma unroll
    for (int v = 0; v < 8; ++v){
      int r    = row0 + wave * 16 + mbase + v;
      int pos  = r & (NSEQ - 1);
      int bidx = r >> 12;
      float s, c; __sincosf((float)pos * invf, &s, &c);
      float a0 = acc[nb][v], a1 = acc[nb + 2][v];
      float r0 = a0 * c - a1 * s;
      float r1 = a1 * c + a0 * s;
      if (!isKV){ r0 *= 0.125f; r1 *= 0.125f; }          // SCALE = 64^-0.5
      size_t base = ((size_t)(bidx * NHEAD + head) * NSEQ + pos) * DH;
      outb[base + d]      = (_Float16)r0;
      outb[base + d + 32] = (_Float16)r1;
    }
  }
}

// ---------------- Kernel 2: local layernorm of kv -> nkv ----------------
__global__ __launch_bounds__(256) void ln_local_kernel(
    const _Float16* __restrict__ kv, const float* __restrict__ g, const float* __restrict__ b,
    _Float16* __restrict__ nkv, int nrows)
{
  int r = blockIdx.x * blockDim.x + threadIdx.x;
  if (r >= nrows) return;
  const _Float16* src = kv + (size_t)r * DH;
  float vb[DH]; float mean = 0.f;
  #pragma unroll
  for (int i = 0; i < DH; ++i){ vb[i] = (float)src[i]; mean += vb[i]; }
  mean *= (1.0f / DH);
  float var = 0.f;
  #pragma unroll
  for (int i = 0; i < DH; ++i){ float d = vb[i] - mean; var += d * d; }
  var *= (1.0f / DH);
  float rs = rsqrtf(var + 1e-5f);
  _Float16* dst = nkv + (size_t)r * DH;
  #pragma unroll
  for (int i = 0; i < DH; ++i) dst[i] = (_Float16)((vb[i] - mean) * rs * g[i] + b[i]);
}

// ---------------- Kernel 3: global pooled keys (gkv) ----------------
__global__ __launch_bounds__(64) void gkv_kernel(
    const _Float16* __restrict__ kv, const float* __restrict__ Wproj,
    const float* __restrict__ g, const float* __restrict__ bb,
    _Float16* __restrict__ gkv)
{
  int blk = blockIdx.x;
  int bh = blk / NSEG, seg = blk % NSEG;
  int t = threadIdx.x;                                  // d = 0..63
  __shared__ float skv[SSEG][DH];
  __shared__ float logit[SSEG];
  __shared__ float gsh[DH];
  const _Float16* base = kv + ((size_t)bh * NSEQ + (size_t)seg * SSEG) * DH;
  #pragma unroll
  for (int s = 0; s < SSEG; ++s) skv[s][t] = (float)base[s * DH + t];
  __syncthreads();
  if (t < SSEG){
    float acc = 0.f;
    #pragma unroll
    for (int d = 0; d < DH; ++d) acc += skv[t][d] * Wproj[d];
    logit[t] = acc;
  }
  __syncthreads();
  float mx = -1e30f;
  #pragma unroll
  for (int s = 0; s < SSEG; ++s) mx = fmaxf(mx, logit[s]);
  float den = 0.f, p[SSEG];
  #pragma unroll
  for (int s = 0; s < SSEG; ++s){ p[s] = __expf(logit[s] - mx); den += p[s]; }
  float inv = 1.0f / den, gv = 0.f;
  #pragma unroll
  for (int s = 0; s < SSEG; ++s) gv += p[s] * inv * skv[s][t];
  gsh[t] = gv;
  __syncthreads();
  float mean = 0.f;
  #pragma unroll
  for (int d = 0; d < DH; ++d) mean += gsh[d];
  mean *= (1.0f / DH);
  float var = 0.f;
  #pragma unroll
  for (int d = 0; d < DH; ++d){ float dd = gsh[d] - mean; var += dd * dd; }
  var *= (1.0f / DH);
  gkv[((size_t)bh * NSEG + seg) * DH + t] = (_Float16)((gv - mean) * rsqrtf(var + 1e-5f) * g[t] + bb[t]);
}

// ---------------- Kernel 4: flash attention over [global | local] keys ----------------
__global__ __launch_bounds__(256) void attn_kernel(
    const _Float16* __restrict__ q, const _Float16* __restrict__ nkv,
    const _Float16* __restrict__ gkv, _Float16* __restrict__ ctx)
{
  const int w  = blockIdx.x;          // 32 windows
  const int bh = blockIdx.y;          // 64
  const int t = threadIdx.x, wave = t >> 5, lane = t & 31;
  const int b = bh >> 4, h = bh & 15;

  __shared__ _Float16 kvch[64][72];   // chunk of 64 keys (K == V); 144B rows: b128-aligned
  __shared__ _Float16 pb[8][16][72];  // per-wave P staging

  // Q fragments for this wave's 16 rows (K = 0..31, 32..63)
  const int qrow0 = w * WWIN + wave * 16;
  v16h qa[2];
  {
    int m = lane & 15, koff = (lane >> 4) * 8;
    const _Float16* qrow = q + ((size_t)bh * NSEQ + qrow0 + m) * DH;
    union { v16h v; unsigned u[8]; } r0, r1;
    #pragma unroll
    for (int v = 0; v < 8; ++v){
      int k = (v < 4) ? (koff + 2*v) : (16 + koff + 2*(v-4));
      r0.u[v] = *(const unsigned*)(qrow + k);
      r1.u[v] = *(const unsigned*)(qrow + 32 + k);
    }
    qa[0] = r0.v; qa[1] = r1.v;
  }

  float mrow[8], lrow[8];
  v8f o[4] = {};
  #pragma unroll
  for (int v = 0; v < 8; ++v){ mrow[v] = -1e30f; lrow[v] = 0.f; }
  const int mbase = 8 * (lane >> 4);

  for (int c = 0; c < 8; ++c){
    __syncthreads();
    { // stage 64 keys: 0..255 global (gkv), 256..511 local (nkv with look-around)
      int kl = t >> 2, c0 = (t & 3) * 16;
      int kidx = c * 64 + kl;
      const _Float16* src = nullptr;
      if (kidx < 256) src = gkv + ((size_t)bh * NSEG + kidx) * DH + c0;
      else {
        int p = (w - 1) * WWIN + (kidx - 256);
        if (p >= 0) src = nkv + ((size_t)bh * NSEQ + p) * DH + c0;
      }
      if (src){
        copy16h_g2l(src, &kvch[kl][c0]);
      } else {
        unsigned* d32 = (unsigned*)&kvch[kl][c0];
        #pragma unroll
        for (int i = 0; i < 8; ++i) d32[i] = 0u;
      }
    }
    async_wait();
    __syncthreads();

    v8f S[4] = {};
    #pragma unroll
    for (int ks = 0; ks < 2; ++ks){
      #pragma unroll
      for (int nb = 0; nb < 4; ++nb){
        v16h bf = load_a_lds<72>(&kvch[0][0], nb * 16, ks * 32);
        S[nb] = __builtin_amdgcn_wmma_f32_16x16x32_f16(false, qa[ks], false, bf, (short)0, S[nb], false, false);
      }
    }
    // masks
    #pragma unroll
    for (int nb = 0; nb < 4; ++nb){
      int j = c * 64 + nb * 16 + (lane & 15);
      #pragma unroll
      for (int v = 0; v < 8; ++v){
        int iabs = w * WWIN + wave * 16 + mbase + v;
        bool ok;
        if (j < 256) ok = iabs >= (j + 1) * SSEG;                 // fully-past segments only
        else { int p = (w - 1) * WWIN + (j - 256); ok = (p >= 0) && (p <= iabs); }
        if (!ok) S[nb][v] = -1e30f;
      }
    }
    // online softmax (rows split across 16-lane halves)
    #pragma unroll
    for (int v = 0; v < 8; ++v){
      float cm = fmaxf(fmaxf(S[0][v], S[1][v]), fmaxf(S[2][v], S[3][v]));
      #pragma unroll
      for (int off = 8; off > 0; off >>= 1) cm = fmaxf(cm, __shfl_xor(cm, off, 16));
      float nm = fmaxf(mrow[v], cm);
      float scale = __expf(mrow[v] - nm);
      float rs = 0.f;
      #pragma unroll
      for (int nb = 0; nb < 4; ++nb){
        float pv = __expf(S[nb][v] - nm);
        S[nb][v] = pv; rs += pv;
      }
      #pragma unroll
      for (int off = 8; off > 0; off >>= 1) rs += __shfl_xor(rs, off, 16);
      lrow[v] = lrow[v] * scale + rs;
      mrow[v] = nm;
      #pragma unroll
      for (int db = 0; db < 4; ++db) o[db][v] *= scale;
    }
    // stage P (C-layout -> A-layout via LDS)
    #pragma unroll
    for (int nb = 0; nb < 4; ++nb)
      #pragma unroll
      for (int v = 0; v < 8; ++v)
        pb[wave][mbase + v][nb * 16 + (lane & 15)] = (_Float16)S[nb][v];
    __syncthreads();
    // O += P @ V
    #pragma unroll
    for (int ks = 0; ks < 2; ++ks){
      v16h pa = load_a_lds<72>(&pb[wave][0][0], 0, ks * 32);
      #pragma unroll
      for (int db = 0; db < 4; ++db){
        v16h vb = load_b_kmajor<72>(&kvch[0][0], db * 16, ks * 32);
        o[db] = __builtin_amdgcn_wmma_f32_16x16x32_f16(false, pa, false, vb, (short)0, o[db], false, false);
      }
    }
  }
  // normalize + store ctx in (b, n, h*64+d) layout
  #pragma unroll
  for (int v = 0; v < 8; ++v){
    float inv = 1.0f / lrow[v];
    int iabs = w * WWIN + wave * 16 + mbase + v;
    size_t base = ((size_t)b * NSEQ + iabs) * DIMM + h * DH;
    #pragma unroll
    for (int db = 0; db < 4; ++db)
      ctx[base + db * 16 + (lane & 15)] = (_Float16)(o[db][v] * inv);
  }
}

// ---------------- Kernel 5: output projection ----------------
__global__ __launch_bounds__(128) void out_proj_kernel(
    const _Float16* __restrict__ ctx, const float* __restrict__ Wout,
    const float* __restrict__ bout, float* __restrict__ out)
{
  const int rowTile = blockIdx.x;  // 256
  const int colTile = blockIdx.y;  // 16
  __shared__ _Float16 As[64][72];
  __shared__ _Float16 Bs[64][72];
  const int t = threadIdx.x, wave = t >> 5, lane = t & 31;
  const int row0 = rowTile * 64, col0 = colTile * 64;
  v8f acc[4] = {};
  for (int kk = 0; kk < DIMM; kk += 64){
    {
      int r = t >> 1, c0 = (t & 1) * 32;
      const _Float16* src = ctx + (size_t)(row0 + r) * DIMM + kk + c0;
      copy16h_g2l(src,      &As[r][c0]);
      copy16h_g2l(src + 16, &As[r][c0 + 16]);
      int kr = t >> 1, n0 = (t & 1) * 32;
      const float* wsrc = Wout + (size_t)(kk + kr) * DIMM + col0 + n0;
      #pragma unroll
      for (int i = 0; i < 32; ++i) Bs[n0 + i][kr] = (_Float16)wsrc[i];
    }
    async_wait();
    __syncthreads();
    #pragma unroll
    for (int ks = 0; ks < 2; ++ks){
      v16h a = load_a_lds<72>(&As[0][0], wave * 16, ks * 32);
      #pragma unroll
      for (int nb = 0; nb < 4; ++nb){
        v16h bf = load_a_lds<72>(&Bs[0][0], nb * 16, ks * 32);
        acc[nb] = __builtin_amdgcn_wmma_f32_16x16x32_f16(false, a, false, bf, (short)0, acc[nb], false, false);
      }
    }
    __syncthreads();
  }
  const int mbase = 8 * (lane >> 4);
  #pragma unroll
  for (int nb = 0; nb < 4; ++nb){
    int colo = col0 + nb * 16 + (lane & 15);
    float bo = bout[colo];
    #pragma unroll
    for (int v = 0; v < 8; ++v){
      int r = row0 + wave * 16 + mbase + v;
      out[(size_t)r * DIMM + colo] = acc[nb][v] + bo;
    }
  }
}

extern "C" void kernel_launch(void* const* d_in, const int* in_sizes, int n_in,
                              void* d_out, int out_size, void* d_ws, size_t ws_size,
                              hipStream_t stream)
{
  const float* x     = (const float*)d_in[0];
  const float* Wq    = (const float*)d_in[1];
  const float* Wkv   = (const float*)d_in[2];
  const float* Wproj = (const float*)d_in[3];
  const float* Wout  = (const float*)d_in[4];
  const float* bout  = (const float*)d_in[5];
  const float* lnlg  = (const float*)d_in[6];
  const float* lnlb  = (const float*)d_in[7];
  const float* lngg  = (const float*)d_in[8];
  const float* lngb  = (const float*)d_in[9];

  char* ws = (char*)d_ws;
  _Float16* qbuf  = (_Float16*)(ws + 0);            // 33,554,432 B
  _Float16* kvbuf = (_Float16*)(ws + 33554432);     // 33,554,432 B
  _Float16* nkvb  = (_Float16*)(ws + 67108864);     // 33,554,432 B
  _Float16* ctxb  = (_Float16*)(ws + 100663296);    // 33,554,432 B
  _Float16* gkvb  = (_Float16*)(ws + 134217728);    //  2,097,152 B

  qkv_rope_kernel<<<dim3(256, 16, 2), 128, 0, stream>>>(x, Wq, Wkv, qbuf, kvbuf);
  ln_local_kernel<<<dim3(1024), 256, 0, stream>>>(kvbuf, lnlg, lnlb, nkvb, BH * NSEQ);
  gkv_kernel<<<dim3(BH * NSEG), 64, 0, stream>>>(kvbuf, Wproj, lngg, lngb, gkvb);
  attn_kernel<<<dim3(32, 64), 256, 0, stream>>>(qbuf, nkvb, gkvb, ctxb);
  out_proj_kernel<<<dim3(256, 16), 128, 0, stream>>>(ctxb, Wout, bout, (float*)d_out);
}
